// SingleGPUMoETorchFFN_2774548873701
// MI455X (gfx1250) — matrix operand order repaired and checked by reading, not compiled, not run on hardware
//
#include <hip/hip_runtime.h>
#include <hip/hip_bf16.h>
#include <math.h>

#define T_TOK 2048
#define DIMSZ 1024
#define HID   4096
#define NE    8
#define MTILE 64
#define HC    32      // HIDDEN chunk per iteration
#define KSLAB 256     // K slab for GEMM1 staging

typedef __attribute__((ext_vector_type(16))) _Float16 v16h;
typedef __attribute__((ext_vector_type(8)))  _Float16 v8h;
typedef __attribute__((ext_vector_type(8)))  float    v8f;

// ---- dynamic LDS layout (bytes), all offsets 16B aligned ----
#define OFF_XA   0
#define SZ_XA    (64 * 1032 * 2)        // x tile  [64][1024+8] f16 = 132096
#define OFF_W1S  (OFF_XA + SZ_XA)
#define SZ_WS1   (32 * 264 * 2)         // w1 slab [32][256+8] f16 = 16896
#define OFF_W3S  (OFF_W1S + SZ_WS1)
#define OFF_W2S  (OFF_W3S + SZ_WS1)
#define SZ_W2S   (1024 * 40 * 2)        // w2 slab [1024][32+8] f16 = 81920
#define OFF_SS   (OFF_W2S + SZ_W2S)
#define SZ_SS    (2 * 64 * 33 * 4)      // S1/S3   [2][64][33] f32 = 16896
#define OFF_HA   (OFF_SS + SZ_SS)
#define SZ_HA    (64 * 40 * 2)          // h tile  [64][32+8] f16 = 5120
#define OFF_TOK  (OFF_HA + SZ_HA)       // 64 ints
#define OFF_WT   (OFF_TOK + 256)        // 64 floats
#define SMEM_BYTES (OFF_WT + 256)       // = 270336 (< 320KB WG limit)

// ---------------- WMMA fragment helpers (f16, 16x16x32) ----------------
// A fragment (16xK=32): lane<16 -> row=lane, K {0..7,16..23}; lane>=16 -> row=lane-16, K {8..15,24..31}
__device__ __forceinline__ v16h frag_a(const _Float16* base, int pitch, int m0, int k0) {
  const int lane = threadIdx.x & 31;
  const _Float16* p = base + (size_t)(m0 + (lane & 15)) * pitch + k0 + ((lane >> 4) << 3);
  v8h lo = *(const v8h*)(p);
  v8h hi = *(const v8h*)(p + 16);
  return __builtin_shufflevector(lo, hi, 0,1,2,3,4,5,6,7,8,9,10,11,12,13,14,15);
}
// B fragment (K=32 x 16): lane<16 -> col=lane, K {0..7,16..23}; lane>=16 -> col=lane-16, K {8..15,24..31}
__device__ __forceinline__ v16h frag_b(const _Float16* base, int pitch, int n0, int k0) {
  const int lane = threadIdx.x & 31;
  const _Float16* p = base + (size_t)(n0 + (lane & 15)) * pitch + k0 + ((lane >> 4) << 3);
  v8h lo = *(const v8h*)(p);
  v8h hi = *(const v8h*)(p + 16);
  return __builtin_shufflevector(lo, hi, 0,1,2,3,4,5,6,7,8,9,10,11,12,13,14,15);
}
__device__ __forceinline__ v8f wmma16(v16h a, v16h b, v8f c) {
  return __builtin_amdgcn_wmma_f32_16x16x32_f16(false, a, false, b, (short)0, c, false, false);
}

// ---------------- gating: scores, top-2, softmax, routing lists ----------------
__global__ __launch_bounds__(256) void moe_gate(const float* __restrict__ x,
                                                const float* __restrict__ gw_g,
                                                int* __restrict__ cnt,
                                                int* __restrict__ toks,
                                                float* __restrict__ wts) {
  __shared__ float gw[NE * DIMSZ];      // 32KB
  for (int u = threadIdx.x; u < NE * DIMSZ; u += 256) gw[u] = gw_g[u];
  __syncthreads();
  const int wid = threadIdx.x >> 5, lane = threadIdx.x & 31;
  const int t = blockIdx.x * 8 + wid;
  const float* xr = x + (size_t)t * DIMSZ;
  float s[NE];
#pragma unroll
  for (int e = 0; e < NE; ++e) s[e] = 0.f;
  for (int k = lane; k < DIMSZ; k += 32) {
    const float xv = xr[k];
#pragma unroll
    for (int e = 0; e < NE; ++e) s[e] += xv * gw[e * DIMSZ + k];
  }
#pragma unroll
  for (int e = 0; e < NE; ++e)
#pragma unroll
    for (int off = 16; off > 0; off >>= 1) s[e] += __shfl_xor(s[e], off, 32);
  if (lane == 0) {
    int i0 = 0; float v0 = s[0];
#pragma unroll
    for (int e = 1; e < NE; ++e) if (s[e] > v0) { v0 = s[e]; i0 = e; }   // ties -> lowest idx
    int i1 = -1; float v1 = -3.4e38f;
#pragma unroll
    for (int e = 0; e < NE; ++e) if (e != i0 && s[e] > v1) { v1 = s[e]; i1 = e; }
    const float w0 = 1.f / (1.f + __expf(v1 - v0));   // softmax over {v0,v1}, v0>=v1
    const float w1 = 1.f - w0;
    int p0 = atomicAdd(&cnt[i0], 1);
    toks[i0 * T_TOK + p0] = t; wts[i0 * T_TOK + p0] = w0;
    int p1 = atomicAdd(&cnt[i1], 1);
    toks[i1 * T_TOK + p1] = t; wts[i1 * T_TOK + p1] = w1;
  }
}

// ---------------- fused expert SwiGLU FFN, routed, WMMA f16 ----------------
__global__ __launch_bounds__(512, 1) void moe_expert(const float* __restrict__ x,
                                                     const float* __restrict__ w1,
                                                     const float* __restrict__ w2,
                                                     const float* __restrict__ w3,
                                                     const int* __restrict__ cnt,
                                                     const int* __restrict__ toks,
                                                     const float* __restrict__ wts,
                                                     float* __restrict__ y) {
  const int e    = blockIdx.x >> 5;         // 32 tiles per expert
  const int tile = blockIdx.x & 31;
  const int m0   = tile * MTILE;
  const int c    = cnt[e];
  if (m0 >= c) return;                      // idle tile -> exit

  extern __shared__ char smem[];
  _Float16* xA   = (_Float16*)(smem + OFF_XA);
  _Float16* w1s  = (_Float16*)(smem + OFF_W1S);
  _Float16* w3s  = (_Float16*)(smem + OFF_W3S);
  _Float16* w2s  = (_Float16*)(smem + OFF_W2S);
  float*    sS   = (float*)   (smem + OFF_SS);
  _Float16* hA   = (_Float16*)(smem + OFF_HA);
  int*      tokS = (int*)     (smem + OFF_TOK);
  float*    wtS  = (float*)   (smem + OFF_WT);

  const int tid  = threadIdx.x;
  const int wid  = tid >> 5;
  const int lane = tid & 31;
  const float* w1e = w1 + (size_t)e * HID * DIMSZ;
  const float* w3e = w3 + (size_t)e * HID * DIMSZ;
  const float* w2e = w2 + (size_t)e * DIMSZ * HID;

  if (tid < MTILE) {
    const int slot = m0 + tid;
    const int cs   = slot < c ? slot : (c - 1);      // clamp tail (weight forced to 0)
    tokS[tid] = toks[e * T_TOK + cs];
    wtS[tid]  = slot < c ? wts[e * T_TOK + cs] : 0.f;
  }
  __syncthreads();

  // gather x rows -> f16 LDS (coalesced float2 per thread)
  for (int u = tid; u < MTILE * (DIMSZ / 2); u += 512) {
    const int row = u >> 9, c2 = u & 511;
    const float2 v = ((const float2*)(x + (size_t)tokS[row] * DIMSZ))[c2];
    _Float16* d = xA + row * 1032 + c2 * 2;
    d[0] = (_Float16)v.x; d[1] = (_Float16)v.y;
  }

  // GEMM1 wave mapping: 16 waves = (mat, mt, nt) over S1/S3 [64 x 32]
  const int g1_mat = wid & 1;
  const int g1_mt  = (wid >> 1) & 3;
  const int g1_nt  = (wid >> 3) & 1;
  // GEMM2 wave mapping: 2 (M halves) x 8 (N strips of 128)
  const int g2_wm = wid & 1;
  const int g2_wn = wid >> 1;

  v8f cacc[2][8];
  {
    v8f z = {};
#pragma unroll
    for (int i = 0; i < 2; ++i)
#pragma unroll
      for (int j = 0; j < 8; ++j) cacc[i][j] = z;
  }

  const _Float16* bsrc = g1_mat ? w3s : w1s;

  for (int hh = 0; hh < HID; hh += HC) {
    // stage w2 slab [1024 d-rows][32 k] -> f16 (read-once from global per WG)
    // + prefetch next chunk's 128B line per row (one chunk ahead of use)
    for (int u = tid; u < DIMSZ * (HC / 2); u += 512) {
      const int row = u >> 4, c2 = u & 15;
      const float* gp = w2e + (size_t)row * HID + hh;
      if ((u & 15) == 0) __builtin_prefetch(gp + HC, 0, 1);   // global_prefetch_b8
      const float2 v = *(const float2*)(gp + c2 * 2);
      _Float16* d = w2s + row * 40 + c2 * 2;
      d[0] = (_Float16)v.x; d[1] = (_Float16)v.y;
    }

    v8f acc1 = {};
    for (int kk = 0; kk < DIMSZ; kk += KSLAB) {
      __syncthreads();                       // prior slab readers done (also orders the w2s stores above before GEMM2)
      // stage w1/w3 slabs [32 hidden rows][256 k] -> f16
      // + prefetch the same position one HIDDEN chunk ahead (streamed exactly once later)
      for (int u = tid; u < HC * (KSLAB / 2); u += 512) {
        const int row = u >> 7, c2 = u & 127;
        const size_t go = (size_t)(hh + row) * DIMSZ + kk + c2 * 2;
        if ((u & 15) == 0) {
          __builtin_prefetch(w1e + go + (size_t)HC * DIMSZ, 0, 1);  // global_prefetch_b8
          __builtin_prefetch(w3e + go + (size_t)HC * DIMSZ, 0, 1);
        }
        const float2 a = *(const float2*)(w1e + go);
        const float2 b = *(const float2*)(w3e + go);
        _Float16* d1 = w1s + row * 264 + c2 * 2;
        _Float16* d3 = w3s + row * 264 + c2 * 2;
        d1[0] = (_Float16)a.x; d1[1] = (_Float16)a.y;
        d3[0] = (_Float16)b.x; d3[1] = (_Float16)b.y;
      }
      __syncthreads();
#pragma unroll
      for (int k0 = 0; k0 < KSLAB; k0 += 32) {
        v16h af = frag_a(xA, 1032, g1_mt * 16, kk + k0);
        v16h bf = frag_b(bsrc, 264, g1_nt * 16, k0);
        acc1 = wmma16(af, bf, acc1);
      }
    }
    // spill S tile to LDS for cross-wave SwiGLU pairing
    {
      const int mrow = g1_mt * 16 + ((lane >> 4) << 3);
      const int ncol = g1_nt * 16 + (lane & 15);
      float* sp = sS + g1_mat * (64 * 33) + mrow * 33 + ncol;
#pragma unroll
      for (int r = 0; r < 8; ++r) sp[r * 33] = acc1[r];
    }
    __syncthreads();
    // h = silu(s1) * s3 -> f16
    for (int u = tid; u < MTILE * HC; u += 512) {
      const int m = u >> 5, n = u & 31;
      const float s1 = sS[m * 33 + n];
      const float s3 = sS[64 * 33 + m * 33 + n];
      hA[m * 40 + n] = (_Float16)((s1 / (1.f + __expf(-s1))) * s3);
    }
    __syncthreads();
    // GEMM2: C[64,1024] += h[64,32] @ w2[:,hh:hh+32]^T
    {
      v16h a0 = frag_a(hA, 40, g2_wm * 32, 0);
      v16h a1 = frag_a(hA, 40, g2_wm * 32 + 16, 0);
#pragma unroll
      for (int nt = 0; nt < 8; ++nt) {
        v16h bf = frag_b(w2s, 40, g2_wn * 128 + nt * 16, 0);
        cacc[0][nt] = wmma16(a0, bf, cacc[0][nt]);
        cacc[1][nt] = wmma16(a1, bf, cacc[1][nt]);
      }
    }
    __syncthreads();                         // protect hA/w2s/sS for next chunk
  }

  // epilogue: y[token] += weight * ye  (exactly 2 contributions/element -> deterministic)
  {
    const int mrow_off = (lane >> 4) << 3;
    const int ncol = lane & 15;
#pragma unroll
    for (int mt = 0; mt < 2; ++mt) {
#pragma unroll
      for (int r = 0; r < 8; ++r) {
        const int m = g2_wm * 32 + mt * 16 + mrow_off + r;
        const float wgt = wtS[m];
        if (wgt != 0.f) {
          const size_t rowoff = (size_t)tokS[m] * DIMSZ + g2_wn * 128 + ncol;
#pragma unroll
          for (int nt = 0; nt < 8; ++nt)
            unsafeAtomicAdd(&y[rowoff + nt * 16], cacc[mt][nt][r] * wgt);
        }
      }
    }
  }
}

extern "C" void kernel_launch(void* const* d_in, const int* in_sizes, int n_in,
                              void* d_out, int out_size, void* d_ws, size_t ws_size,
                              hipStream_t stream) {
  const float* x  = (const float*)d_in[0];
  const float* gw = (const float*)d_in[1];
  const float* w1 = (const float*)d_in[2];
  const float* w2 = (const float*)d_in[3];
  const float* w3 = (const float*)d_in[4];
  float* y = (float*)d_out;

  int*   cnt  = (int*)d_ws;                                   // 8 counters (32B slot)
  int*   toks = (int*)((char*)d_ws + 32);                     // [8][2048] token ids
  float* wts  = (float*)((char*)d_ws + 32 + NE * T_TOK * 4);  // [8][2048] weights

  hipMemsetAsync(d_out, 0, (size_t)out_size * sizeof(float), stream);
  hipMemsetAsync(cnt, 0, 32, stream);

  moe_gate<<<T_TOK / 8, 256, 0, stream>>>(x, gw, cnt, toks, wts);

  hipFuncSetAttribute((const void*)moe_expert,
                      hipFuncAttributeMaxDynamicSharedMemorySize, SMEM_BYTES);
  moe_expert<<<NE * (T_TOK / MTILE), 512, SMEM_BYTES, stream>>>(
      x, w1, w2, w3, cnt, toks, wts, y);
}
